// MemoryGraph_85581518340291
// MI455X (gfx1250) — matrix operand order repaired and checked by reading, not compiled, not run on hardware
//
#include <hip/hip_runtime.h>
#include <hip/hip_bf16.h>

#define N_NEUR 4096
#define KCONN  32
#define DDIM   64
#define HSD    256
#define BSZ    8
#define TSTEPS 32
#define MODIN  225
#define MODHID 64
#define MODOUT 97
#define SW1_LD 193   // state_w1 row length (3*D+1)
#define MW1_LD 192   // msg_w1 row length (3*D)

typedef __attribute__((ext_vector_type(16))) _Float16 v16h;
typedef __attribute__((ext_vector_type(8)))  _Float16 v8h;
typedef __attribute__((ext_vector_type(8)))  float    v8f;

// CDNA5 hardware tanh (TRANS op, co-executes with WMMA) instead of branchy libm tanhf.
static __device__ __forceinline__ float fast_tanh(float x) {
#if __has_builtin(__builtin_amdgcn_tanhf)
  return __builtin_amdgcn_tanhf(x);
#else
  float r;
  asm volatile("v_tanh_f32 %0, %1" : "=v"(r) : "v"(x));
  return r;
#endif
}
static __device__ __forceinline__ float fast_sigmoid(float x) {
  return 0.5f + 0.5f*fast_tanh(0.5f*x);
}

static __device__ __forceinline__ v16h ld_pair(const _Float16* p0, const _Float16* p1) {
  union { v16h v; v8h p[2]; } u;
  u.p[0] = *(const v8h*)p0;
  u.p[1] = *(const v8h*)p1;
  return u.v;
}

static __device__ __forceinline__ v8f wmma_f16(v16h a, v16h b, v8f c) {
  // D = A(16x32 f16) x B(32x16 f16) + C(16x16 f32)
  return __builtin_amdgcn_wmma_f32_16x16x32_f16(false, a, false, b, (short)0, c, false, false);
}

// A fragment (16x32 f16 tile) from LDS: row m, row-stride `stride` halves, k-base kbase.
static __device__ __forceinline__ v16h afrag(const _Float16* s, int m, int stride, int kbase, int half) {
  const _Float16* p = s + m*stride + kbase + half*8;
  return ld_pair(p, p + 16);
}

// B fragment from transposed f16 weights WT[k][n] (row length nlen):
// column n = lane%16; registers hold k = 8*half+s (s<8), 16+8*half+(s-8) (s>=8).
static __device__ __forceinline__ v16h bfrag(const _Float16* WT, int nlen, int ct, int kbase, int lane) {
  const _Float16* p = WT + (kbase + 8*(lane >> 4))*nlen + ct*16 + (lane & 15);
  v16h v;
  #pragma unroll
  for (int s = 0; s < 8; ++s) {
    v[s]     = p[s*nlen];
    v[s + 8] = p[(s + 16)*nlen];
  }
  return v;
}

// ---------------------------------------------------------------------------
// P1: convert the 4 shared weight matrices to transposed f16, copy h/msg state
// into workspace (inputs must never be mutated).
// ---------------------------------------------------------------------------
__global__ __launch_bounds__(256) void prep_kernel(
    const float* __restrict__ sw1, const float* __restrict__ sw2,
    const float* __restrict__ mw1, const float* __restrict__ mw2,
    const float* __restrict__ h0,  const float* __restrict__ msg0,
    _Float16* __restrict__ WB1, _Float16* __restrict__ WB2,
    _Float16* __restrict__ WB3, _Float16* __restrict__ WB4,
    float* __restrict__ h_ws, float* __restrict__ msg_a) {
  int i = blockIdx.x*256 + threadIdx.x;
  if (i < 16384) {
    // WB1/WB3: (K=64, N=256) transposed from (256 x rowlen) weight slices [:, :64]
    int k = i >> 8, n = i & 255;
    WB1[i] = (_Float16)sw1[n*SW1_LD + k];
    WB3[i] = (_Float16)mw1[n*MW1_LD + k];
    // WB2/WB4: (K=256, N=64) transposed from (64 x 256)
    int k2 = i >> 6, n2 = i & 63;
    WB2[i] = (_Float16)sw2[n2*256 + k2];
    WB4[i] = (_Float16)mw2[n2*256 + k2];
  }
  if (i < BSZ*N_NEUR*DDIM) {
    h_ws[i]  = h0[i];
    msg_a[i] = msg0[i];
  }
}

// ---------------------------------------------------------------------------
// P2: per-neuron modulation MLP + precompute of state_const / msg_const,
// w_sig, sigmoid(decay). One workgroup per neuron, per-neuron weights staged
// once in LDS (f16) and reused across the 8 batch rows.
// ---------------------------------------------------------------------------
__global__ __launch_bounds__(256) void mod_kernel(
    const float* __restrict__ hebb, const float* __restrict__ h0,
    const float* __restrict__ dl0,  const float* __restrict__ prim0,
    const float* __restrict__ wconn, const float* __restrict__ nid,
    const float* __restrict__ modw1, const float* __restrict__ modb1,
    const float* __restrict__ modw2, const float* __restrict__ modb2,
    const float* __restrict__ sw1, const float* __restrict__ sb1,
    const float* __restrict__ mw1, const float* __restrict__ mb1,
    float* __restrict__ w_sig, float* __restrict__ dsig,
    float* __restrict__ state_const, float* __restrict__ msg_const) {
  __shared__ _Float16 w1s[MODHID*MODIN];
  __shared__ _Float16 w2s[MODHID*MODOUT];
  __shared__ float x[MODIN];
  __shared__ float hid[MODHID];
  __shared__ float delta[MODOUT];
  __shared__ float sprim[DDIM];
  __shared__ float red[DDIM];
  __shared__ float sdl, srms;

  const int n = blockIdx.x;
  const int t = threadIdx.x;
  for (int i = t; i < MODHID*MODIN;  i += 256) w1s[i] = (_Float16)modw1[n*(MODHID*MODIN) + i];
  for (int i = t; i < MODHID*MODOUT; i += 256) w2s[i] = (_Float16)modw2[n*(MODHID*MODOUT) + i];
  __syncthreads();

  for (int b = 0; b < BSZ; ++b) {
    const int row = b*N_NEUR + n;
    // mod_input = [hebbian(32) | h(64) | decay(1) | prim(64) | nid(64)]
    for (int i = t; i < MODIN; i += 256) {
      float v;
      if (i < 32)       v = hebb[row*KCONN + i];
      else if (i < 96)  v = h0[row*DDIM + (i-32)];
      else if (i == 96) v = dl0[row];
      else if (i < 161) v = prim0[row*DDIM + (i-97)];
      else              v = nid[n*DDIM + (i-161)];
      x[i] = v;
    }
    __syncthreads();
    if (t < MODHID) {
      float s = modb1[n*MODHID + t];
      const _Float16* wr = &w1s[t*MODIN];
      for (int i = 0; i < MODIN; ++i) s += (float)wr[i] * x[i];
      hid[t] = fast_tanh(s);
    }
    __syncthreads();
    if (t < MODOUT) {
      float s = modb2[n*MODOUT + t];
      for (int j = 0; j < MODHID; ++j) s += hid[j] * (float)w2s[j*MODOUT + t];
      delta[t] = s;
    }
    __syncthreads();
    if (t < KCONN) {
      float w = wconn[row*KCONN + t] + delta[t];
      w_sig[row*KCONN + t] = fast_sigmoid(w);
    }
    if (t == KCONN) {
      float dl = dl0[row] + delta[KCONN];
      sdl = dl;
      dsig[row] = fast_sigmoid(dl);
    }
    if (t < DDIM) {
      float p = prim0[row*DDIM + t] + delta[KCONN+1+t];
      sprim[t] = p; red[t] = p*p;
    }
    __syncthreads();
    if (t == 0) {
      float s = 0.f;
      for (int i = 0; i < DDIM; ++i) s += red[i];
      srms = rsqrtf(s*(1.0f/DDIM) + 1e-6f);
    }
    __syncthreads();
    if (t < DDIM) sprim[t] *= srms;
    __syncthreads();
    {
      const float dl = sdl;
      const float* wr = sw1 + t*SW1_LD;
      const float* mr = mw1 + t*MW1_LD;
      float sc = sb1[t] + dl*wr[192];
      float mc = mb1[t];
      for (int d = 0; d < DDIM; ++d) {
        float pd = sprim[d], nd = x[161+d];
        sc += pd*wr[64+d] + nd*wr[128+d];
        mc += pd*mr[64+d] + nd*mr[128+d];
      }
      state_const[row*HSD + t] = sc;
      msg_const[row*HSD + t]  = mc;
    }
    __syncthreads();
  }
}

// ---------------------------------------------------------------------------
// Per-timestep kernel. 256 threads = 8 wave32; 64 rows per block (4 tiles of
// 16); all four GEMMs via v_wmma_f32_16x16x32_f16. B1/B3 fragments are
// VGPR-resident; B2/B4 fragments are reloaded per tile from L2-resident
// weights to stay under the 256-VGPR direct window.
// ---------------------------------------------------------------------------
__global__ __launch_bounds__(256) void step_kernel(
    const float* __restrict__ cc, const int* __restrict__ conn,
    const float* __restrict__ sb2, const float* __restrict__ mb2,
    const float* __restrict__ nid,
    const float* __restrict__ state_const, const float* __restrict__ msg_const,
    const float* __restrict__ w_sig, const float* __restrict__ dsig,
    float* __restrict__ h_ws,
    const float* __restrict__ msg_in, float* __restrict__ msg_out,
    const _Float16* __restrict__ WB1, const _Float16* __restrict__ WB2,
    const _Float16* __restrict__ WB3, const _Float16* __restrict__ WB4,
    float* __restrict__ out, int tstep) {
  __shared__ _Float16 sBig[16*HSD];    // hid / mh staging (16 x 256 f16)
  __shared__ _Float16 sSmall[16*DDIM]; // recv / h staging (16 x 64 f16)
  __shared__ float sD[16];

  const int tid  = threadIdx.x;
  const int wid  = tid >> 5;
  const int lane = tid & 31;
  const int half = lane >> 4;
  const int ln   = lane & 15;
  const int ct0  = 2*wid;

  // Persistent B fragments for GEMM1/GEMM3 only (64 VGPRs).
  v16h B1[2][2], B3[2][2];
  #pragma unroll
  for (int ci = 0; ci < 2; ++ci)
    #pragma unroll
    for (int kt = 0; kt < 2; ++kt) {
      B1[ci][kt] = bfrag(WB1, 256, ct0+ci, kt*32, lane);
      B3[ci][kt] = bfrag(WB3, 256, ct0+ci, kt*32, lane);
    }
  // GEMM2 (waves 0-3) / GEMM4 (waves 4-7) weight base for per-tile reload.
  const _Float16* WB24 = (wid < 4) ? WB2 : WB4;
  const int ct24 = (wid < 4) ? wid : (wid - 4);

  for (int rt = 0; rt < 4; ++rt) {
    const int rb = blockIdx.x*64 + rt*16;   // global row base (b*N + n)
    const int b  = rb >> 12;
    const int nb = rb & (N_NEUR-1);

    // ---- recv gather: recv[r][d] = sum_k w_sig[k] * msg_prev[conn[k]][d] (+inject)
    {
      const int r = tid >> 4, cg = tid & 15;
      const int n = nb + r;
      const int row = rb + r;
      float a0 = 0.f, a1 = 0.f, a2 = 0.f, a3 = 0.f;
      const float* msrc = msg_in + (b << 12)*DDIM;
      const int*   cp = conn + n*KCONN;
      const float* wp = w_sig + row*KCONN;
      for (int k = 0; k < KCONN; ++k) {
        const int c = cp[k];
        const float w = wp[k];
        const float4 mm = *(const float4*)(msrc + c*DDIM + cg*4);
        a0 += w*mm.x; a1 += w*mm.y; a2 += w*mm.z; a3 += w*mm.w;
      }
      if ((n & 127) < 4) {              // inject_indices = {s*128+0..3}
        const int s = n >> 7;
        const float4 c4 = *(const float4*)(cc + (b*TSTEPS + tstep)*2048 + s*DDIM + cg*4);
        a0 += c4.x; a1 += c4.y; a2 += c4.z; a3 += c4.w;
      }
      _Float16* dp = sSmall + r*DDIM + cg*4;
      dp[0] = (_Float16)a0; dp[1] = (_Float16)a1;
      dp[2] = (_Float16)a2; dp[3] = (_Float16)a3;
      if (tid < 16) sD[tid] = dsig[rb + tid];
    }
    __syncthreads();

    // ---- GEMM1: hid = tanh(recv @ Wst_in^T + state_const)   (16x64)x(64x256)
    {
      const v16h a0 = afrag(sSmall, ln, DDIM, 0,  half);
      const v16h a1 = afrag(sSmall, ln, DDIM, 32, half);
      #pragma unroll
      for (int ci = 0; ci < 2; ++ci) {
        v8f acc = {0,0,0,0,0,0,0,0};
        acc = wmma_f16(a0, B1[ci][0], acc);
        acc = wmma_f16(a1, B1[ci][1], acc);
        const int col = (ct0+ci)*16 + ln;
        #pragma unroll
        for (int g = 0; g < 8; ++g) {
          const int m = g + 8*half;
          const float v = fast_tanh(acc[g] + state_const[(rb+m)*HSD + col]);
          sBig[m*HSD + col] = (_Float16)v;
        }
      }
    }
    __syncthreads();

    // ---- GEMM2 (waves 0-3): upd = tanh(hid @ Wst2^T + b2); h = d*h + (1-d)*upd
    if (wid < 4) {
      v8f acc = {0,0,0,0,0,0,0,0};
      #pragma unroll
      for (int kt = 0; kt < 8; ++kt) {
        const v16h a = afrag(sBig, ln, HSD, kt*32, half);
        const v16h bw = bfrag(WB24, 64, ct24, kt*32, lane);
        acc = wmma_f16(a, bw, acc);
      }
      const int col = ct24*16 + ln;
      const float bb = sb2[col];
      #pragma unroll
      for (int g = 0; g < 8; ++g) {
        const int m = g + 8*half;
        const float upd = fast_tanh(acc[g] + bb);
        const float dd = sD[m];
        const int idx = (rb+m)*DDIM + col;
        const float hn = dd*h_ws[idx] + (1.0f - dd)*upd;
        h_ws[idx] = hn;
        sSmall[m*DDIM + col] = (_Float16)hn;
      }
    }
    __syncthreads();

    // ---- GEMM3: mh = tanh(h @ Wmh^T + msg_const)
    {
      const v16h a0 = afrag(sSmall, ln, DDIM, 0,  half);
      const v16h a1 = afrag(sSmall, ln, DDIM, 32, half);
      #pragma unroll
      for (int ci = 0; ci < 2; ++ci) {
        v8f acc = {0,0,0,0,0,0,0,0};
        acc = wmma_f16(a0, B3[ci][0], acc);
        acc = wmma_f16(a1, B3[ci][1], acc);
        const int col = (ct0+ci)*16 + ln;
        #pragma unroll
        for (int g = 0; g < 8; ++g) {
          const int m = g + 8*half;
          const float v = fast_tanh(acc[g] + msg_const[(rb+m)*HSD + col]);
          sBig[m*HSD + col] = (_Float16)v;
        }
      }
    }
    __syncthreads();

    // ---- GEMM4 (waves 4-7): msg = tanh(mh @ Wm2^T + b2) + nid; readout
    if (wid >= 4) {
      v8f acc = {0,0,0,0,0,0,0,0};
      #pragma unroll
      for (int kt = 0; kt < 8; ++kt) {
        const v16h a = afrag(sBig, ln, HSD, kt*32, half);
        const v16h bw = bfrag(WB24, 64, ct24, kt*32, lane);
        acc = wmma_f16(a, bw, acc);
      }
      const int col = ct24*16 + ln;
      const float bb = mb2[col];
      float vals[8];
      #pragma unroll
      for (int g = 0; g < 8; ++g) {
        const int m = g + 8*half;
        const float mv = fast_tanh(acc[g] + bb) + nid[(nb+m)*DDIM + col];
        msg_out[(rb+m)*DDIM + col] = mv;
        vals[g] = mv;
      }
      // readout rows n%128 in [124,127] -> tile base n%128 == 112, rows m=12..15
      if (((nb & 127) == 112) && half == 1) {
        const float ro = 0.25f*(vals[4]+vals[5]+vals[6]+vals[7]);
        out[(b*TSTEPS + tstep)*2048 + (nb >> 7)*DDIM + col] = ro;
      }
    }
    __syncthreads();
  }
}

// ---------------------------------------------------------------------------
extern "C" void kernel_launch(void* const* d_in, const int* in_sizes, int n_in,
                              void* d_out, int out_size, void* d_ws, size_t ws_size,
                              hipStream_t stream) {
  const float* cc    = (const float*)d_in[0];
  const float* h0    = (const float*)d_in[1];
  const float* msg0  = (const float*)d_in[2];
  const float* wconn = (const float*)d_in[3];
  const float* dl0   = (const float*)d_in[4];
  const float* prim0 = (const float*)d_in[5];
  const float* hebb  = (const float*)d_in[6];
  const float* sw1   = (const float*)d_in[7];
  const float* sb1   = (const float*)d_in[8];
  const float* sw2   = (const float*)d_in[9];
  const float* sb2   = (const float*)d_in[10];
  const float* mw1   = (const float*)d_in[11];
  const float* mb1   = (const float*)d_in[12];
  const float* mw2   = (const float*)d_in[13];
  const float* mb2   = (const float*)d_in[14];
  const float* modw1 = (const float*)d_in[15];
  const float* modb1 = (const float*)d_in[16];
  const float* modw2 = (const float*)d_in[17];
  const float* modb2 = (const float*)d_in[18];
  const float* nid   = (const float*)d_in[19];
  const int*   conn  = (const int*)d_in[20];
  float* outp = (float*)d_out;

  // workspace partition (all freshly written each call; never reads stale state)
  float* ws = (float*)d_ws;
  float* state_const = ws; ws += (size_t)BSZ*N_NEUR*HSD;   // 8.4M f
  float* msg_const   = ws; ws += (size_t)BSZ*N_NEUR*HSD;   // 8.4M f
  float* h_ws        = ws; ws += (size_t)BSZ*N_NEUR*DDIM;  // 2.1M f
  float* msg_a       = ws; ws += (size_t)BSZ*N_NEUR*DDIM;
  float* msg_b       = ws; ws += (size_t)BSZ*N_NEUR*DDIM;
  float* w_sigp      = ws; ws += (size_t)BSZ*N_NEUR*KCONN; // 1.0M f
  float* dsigp       = ws; ws += (size_t)BSZ*N_NEUR;
  _Float16* WB1 = (_Float16*)ws;       // (64,256)  state_w_in^T
  _Float16* WB2 = WB1 + 64*256;        // (256,64)  state_w2^T
  _Float16* WB3 = WB2 + 256*64;        // (64,256)  msg_w_h^T
  _Float16* WB4 = WB3 + 64*256;        // (256,64)  msg_w2^T

  prep_kernel<<<(BSZ*N_NEUR*DDIM + 255)/256, 256, 0, stream>>>(
      sw1, sw2, mw1, mw2, h0, msg0, WB1, WB2, WB3, WB4, h_ws, msg_a);

  mod_kernel<<<N_NEUR, 256, 0, stream>>>(
      hebb, h0, dl0, prim0, wconn, nid,
      modw1, modb1, modw2, modb2,
      sw1, sb1, mw1, mb1,
      w_sigp, dsigp, state_const, msg_const);

  for (int t = 0; t < TSTEPS; ++t) {
    const float* min_ = (t & 1) ? msg_b : msg_a;
    float*       mout = (t & 1) ? msg_a : msg_b;
    step_kernel<<<(BSZ*N_NEUR)/64, 256, 0, stream>>>(
        cc, conn, sb2, mb2, nid,
        state_const, msg_const, w_sigp, dsigp,
        h_ws, min_, mout,
        WB1, WB2, WB3, WB4, outp, t);
  }
}
